// EventMemoryCell_75247827026352
// MI455X (gfx1250) — matrix operand order repaired and checked by reading, not compiled, not run on hardware
//
#include <hip/hip_runtime.h>
#include <hip/hip_bf16.h>
#include <cstdint>
#include <cstddef>

// ---------------------------------------------------------------------------
// EventMemoryCell for MI455X (gfx1250, wave32, WMMA).
//   B=2048, D=256, H=256, N=64, F=513.
// Stage 1 (select_kernel): q/qk/v GEMVs + slot selection + state updates.
//   Uses sims[b,n] = slots[b,n] . (Wk^T q[b]) (avoids the big keys GEMM).
//   Also emits a bf16-packed mem_seq x-part into workspace for stage 2.
// Stage 2 (lstm_kernel): 64-step LSTM, batch-tiled 16 rows/block,
//   v_wmma_f32_16x16x32_bf16 over K=768 (slots|cum|h), delta as rank-1 VALU.
//   X staged to LDS via gfx1250 async load-to-LDS (ASYNCcnt) when available.
// ---------------------------------------------------------------------------

typedef __bf16 bf16_t;
typedef __attribute__((ext_vector_type(16))) __bf16 v16bf;
typedef __attribute__((ext_vector_type(8)))  __bf16 v8bf;
typedef __attribute__((ext_vector_type(8)))  float  v8f;
typedef __attribute__((ext_vector_type(4)))  int    v4i_t;

#define B_SZ   2048
#define D_SZ   256
#define H_SZ   256
#define N_SLOT 64
#define KCAT   768    // 512 (slots|cum) + 256 (h)
#define XPITCH 776    // padded LDS row pitch (bf16 elems), keeps 16B alignment

#if defined(__has_builtin)
#  if __has_builtin(__builtin_amdgcn_global_load_async_to_lds_b128)
#    define HAVE_ASYNC_LDS 1
#  endif
#endif
#ifndef HAVE_ASYNC_LDS
#  define HAVE_ASYNC_LDS 0
#endif

__device__ __forceinline__ float sigm(float x) { return 1.0f / (1.0f + __expf(-x)); }

__device__ __forceinline__ v8f wmma_bf16(v16bf a, v16bf b, v8f c) {
  // D = A(16x32 bf16) * B(32x16 bf16) + C(16x16 f32)
  return __builtin_amdgcn_wmma_f32_16x16x32_bf16(false, a, false, b, (short)0, c,
                                                 false, false);
}

// ---------------------------------------------------------------------------
// Weight prep: Wcat[row][0:512] = bf16(W_ih[row][0:512]);
//              Wcat[row][512:768] = bf16(W_hh[row][0:256]);
//              Wd[row] = W_ih[row][512] (delta_t column); bias = b_ih + b_hh.
// ---------------------------------------------------------------------------
__global__ __launch_bounds__(256) void prep_weights(
    const float* __restrict__ W_ih, const float* __restrict__ W_hh,
    const float* __restrict__ b_ih, const float* __restrict__ b_hh,
    bf16_t* __restrict__ Wcat, float* __restrict__ Wd, float* __restrict__ biasc)
{
  const int row = blockIdx.x;     // 0..1023 (4H)
  const int tid = threadIdx.x;
  const float* wr = W_ih + (size_t)row * 513;
  const float* hr = W_hh + (size_t)row * 256;
  for (int k = tid; k < 512; k += 256)
    Wcat[(size_t)row * KCAT + k] = (bf16_t)wr[k];
  for (int k = tid; k < 256; k += 256)
    Wcat[(size_t)row * KCAT + 512 + k] = (bf16_t)hr[k];
  if (tid == 0) {
    Wd[row]    = wr[512];
    biasc[row] = b_ih[row] + b_hh[row];
  }
}

// ---------------------------------------------------------------------------
// Selection + state update. One block per batch row, 256 threads.
// Optionally packs bf16 mem_seq x-part [slots|cum] into xbf_w (ws).
// ---------------------------------------------------------------------------
__global__ __launch_bounds__(256) void select_kernel(
    const float* __restrict__ x_t,  const float* __restrict__ slots_in,
    const float* __restrict__ cum_in, const float* __restrict__ dt_in,
    const int* __restrict__ filled_in,
    const float* __restrict__ Wq, const float* __restrict__ Wk,
    const float* __restrict__ Wv, const float* __restrict__ bv,
    float* __restrict__ slots_out, float* __restrict__ cum_out,
    float* __restrict__ dt_out,    float* __restrict__ filled_out,
    bf16_t* __restrict__ xbf_w)
{
  __shared__ float xs[256], qs[256], qks[256], vs[256], sims[64];
  __shared__ int sidx;
  const int tid = threadIdx.x;
  const int b   = blockIdx.x;

  xs[tid] = x_t[(size_t)b * 256 + tid];
  __syncthreads();

  // q = x . Wq^T   (thread tid -> q[tid])
  float aq = 0.f;
  for (int d = 0; d < 256; ++d) aq += xs[d] * Wq[(size_t)tid * 256 + d];
  qs[tid] = aq;
  __syncthreads();

  // qk = Wk^T q  (qk[d] = sum_h q[h] * Wk[h][d]) -- coalesced across threads
  float ak = 0.f;
  for (int h = 0; h < 256; ++h) ak += qs[h] * Wk[(size_t)h * 256 + tid];
  qks[tid] = ak;
  // v = x . Wv^T + bv
  float av = bv[tid];
  for (int d = 0; d < 256; ++d) av += xs[d] * Wv[(size_t)tid * 256 + d];
  vs[tid] = av;
  __syncthreads();

  // sims[n] = slots[b,n] . qk
  if (tid < 64) {
    const float* sr = slots_in + ((size_t)b * N_SLOT + tid) * D_SZ;
    float s = 0.f;
    for (int d = 0; d < 256; ++d) s += sr[d] * qks[d];
    sims[tid] = s;
  }
  __syncthreads();

  if (tid == 0) {
    int idx = -1;                              // first empty slot wins
    for (int nI = 0; nI < 64; ++nI)
      if (!filled_in[(size_t)b * 64 + nI]) { idx = nI; break; }
    if (idx < 0) {                             // else first argmax of sims
      float best = sims[0]; idx = 0;
      for (int nI = 1; nI < 64; ++nI)
        if (sims[nI] > best) { best = sims[nI]; idx = nI; }
    }
    sidx = idx;
  }
  __syncthreads();
  const int idx = sidx;

  for (int nI = 0; nI < 64; ++nI) {
    size_t o = ((size_t)b * N_SLOT + nI) * D_SZ + tid;
    float so = (nI == idx) ? vs[tid] : slots_in[o];
    float co = (nI == idx) ? xs[tid] : cum_in[o] + xs[tid];
    slots_out[o] = so;
    cum_out[o]   = co;
    if (xbf_w) {
      size_t xo = ((size_t)b * N_SLOT + nI) * 512;
      xbf_w[xo + tid]       = (bf16_t)so;
      xbf_w[xo + 256 + tid] = (bf16_t)co;
    }
  }
  if (tid < 64) {
    size_t o = (size_t)b * 64 + tid;
    dt_out[o]     = (tid == idx) ? 0.f : dt_in[o] + 1.f;
    filled_out[o] = (tid == idx) ? 1.f : (filled_in[o] ? 1.f : 0.f);
  }
}

// ---------------------------------------------------------------------------
// LSTM over N=64 steps. 128 blocks x 256 threads; 16 batch rows per block.
// Per step, per wave: 8 interleaved 16x16 output tiles (stride-8 => each wave
// owns matching i/f/g/o columns), 24 bf16 WMMA k-chunks each.
// XBF=true: stage pre-packed bf16 x-part via async global->LDS copies.
// ---------------------------------------------------------------------------
template <bool XBF>
__global__ __launch_bounds__(256) void lstm_kernel(
    const float* __restrict__ slots, const float* __restrict__ cum,
    const bf16_t* __restrict__ xbf,  const float* __restrict__ dt,
    const bf16_t* __restrict__ Wcat, const float* __restrict__ Wd,
    const float* __restrict__ biasc, float* __restrict__ h_out)
{
  __shared__ bf16_t Xs[16][XPITCH];   // cols 0..511: slots|cum ; 512..767: h
  __shared__ float dLDS[16];

  const int tid  = threadIdx.x;
  const int b0   = blockIdx.x * 16;
  const int w    = tid >> 5;          // wave 0..7
  const int lane = tid & 31;
  const int nn   = lane & 15;         // N index within a 16x16 tile
  const int hi   = lane >> 4;         // lane-half selector

  // h0 = 0
  for (int i = tid; i < 16 * 256; i += 256)
    Xs[i >> 8][512 + (i & 255)] = (bf16_t)0.0f;

  float creg[2][8];                   // c state: [jt][r] ; m = r + 8*hi
  #pragma unroll
  for (int jt = 0; jt < 2; ++jt)
    #pragma unroll
    for (int r = 0; r < 8; ++r) creg[jt][r] = 0.0f;

  for (int n = 0; n < N_SLOT; ++n) {
    __syncthreads();                  // prior step's reads done
    // Stage x-part of this step's input (16 rows x 512 bf16 cols).
    if (XBF) {
      // Pre-packed bf16: pure byte copy global -> LDS, 16B granules.
      for (int g = tid; g < 16 * 64; g += 256) {
        int row = g >> 6, c8 = (g & 63) << 3;
        const bf16_t* src = xbf + ((size_t)(b0 + row) * N_SLOT + n) * 512 + c8;
        bf16_t* dst = &Xs[row][c8];
#if HAVE_ASYNC_LDS
        __builtin_amdgcn_global_load_async_to_lds_b128(
            (__attribute__((address_space(1))) v4i_t*)src,
            (__attribute__((address_space(3))) v4i_t*)dst,
            0, 0);
#else
        *(v8bf*)dst = *(const v8bf*)src;
#endif
      }
#if HAVE_ASYNC_LDS
      asm volatile("s_wait_asynccnt 0x0" ::: "memory");
#endif
    } else {
      for (int i = tid; i < 16 * 512; i += 256) {
        int row = i >> 9, col = i & 511;
        size_t base = ((size_t)(b0 + row) * N_SLOT + n) * D_SZ;
        float v = (col < 256) ? slots[base + col] : cum[base + (col - 256)];
        Xs[row][col] = (bf16_t)v;
      }
    }
    if (tid < 16) dLDS[tid] = dt[(size_t)(b0 + tid) * N_SLOT + n];
    __syncthreads();

    // Init accumulators with fused bias (same for every row of a column).
    v8f acc[8];
    #pragma unroll
    for (int tt = 0; tt < 8; ++tt) {
      int col0 = (w + 8 * tt) * 16;
      float bc = biasc[col0 + nn];
      #pragma unroll
      for (int r = 0; r < 8; ++r) acc[tt][r] = bc;
    }

    // K loop: 24 chunks of 32 bf16 (slots|cum|h), one A frag reused 8x.
    for (int kc = 0; kc < 24; ++kc) {
      const int k0 = kc * 32;
      union { v16bf v; v8bf h[2]; } afrag;
      // A layout (16-bit 16x32): lane m = lane&15; k = k0 + 16*hi + {0..7}
      // for regs 0..3 and k0 + 16*hi + 8 + {0..7} for regs 4..7.
      afrag.h[0] = *(const v8bf*)(&Xs[nn][k0 + 16 * hi]);
      afrag.h[1] = *(const v8bf*)(&Xs[nn][k0 + 16 * hi + 8]);
      #pragma unroll
      for (int tt = 0; tt < 8; ++tt) {
        int col0 = (w + 8 * tt) * 16;
        // B layout (16-bit 32x16): lane n = lane&15; k = 16*hi + {0..15},
        // contiguous 32B run of Wcat row (col0+nn).
        v16bf bfrag = *(const v16bf*)(Wcat + (size_t)(col0 + nn) * KCAT + k0 + 16 * hi);
        acc[tt] = wmma_bf16(afrag.v, bfrag, acc[tt]);
      }
    }

    // delta_t contribution: rank-1 update with W_ih column 512.
    #pragma unroll
    for (int tt = 0; tt < 8; ++tt) {
      int col0 = (w + 8 * tt) * 16;
      float wd = Wd[col0 + nn];
      #pragma unroll
      for (int r = 0; r < 8; ++r) acc[tt][r] += dLDS[r + 8 * hi] * wd;
    }

    // Gates. Tile pairs: (0,1)=i, (2,3)=f, (4,5)=g, (6,7)=o; jt selects the
    // 16-column group (j = w*16+nn + 128*jt), identical across gates.
    float hreg[2][8];
    #pragma unroll
    for (int jt = 0; jt < 2; ++jt) {
      #pragma unroll
      for (int r = 0; r < 8; ++r) {
        float zi = acc[0 + jt][r], zf = acc[2 + jt][r];
        float zg = acc[4 + jt][r], zo = acc[6 + jt][r];
        float c  = creg[jt][r];
        c = sigm(zf) * c + sigm(zi) * tanhf(zg);
        float h = sigm(zo) * tanhf(c);
        creg[jt][r] = c;
        hreg[jt][r] = h;
      }
    }
    __syncthreads();                  // everyone done reading h_prev
    #pragma unroll
    for (int jt = 0; jt < 2; ++jt) {
      int j = w * 16 + nn + 128 * jt;
      #pragma unroll
      for (int r = 0; r < 8; ++r) {
        Xs[r + 8 * hi][512 + j] = (bf16_t)hreg[jt][r];
        if (n == N_SLOT - 1)
          h_out[(size_t)(b0 + r + 8 * hi) * H_SZ + j] = hreg[jt][r];
      }
    }
  }
}

// ---------------------------------------------------------------------------
extern "C" void kernel_launch(void* const* d_in, const int* in_sizes, int n_in,
                              void* d_out, int out_size, void* d_ws, size_t ws_size,
                              hipStream_t stream)
{
  const float* x_t    = (const float*)d_in[0];
  /* d_in[1] = h_mem_prev (unused by reference) */
  const float* slots  = (const float*)d_in[2];
  const float* cum    = (const float*)d_in[3];
  const float* dt     = (const float*)d_in[4];
  const int*   filled = (const int*)  d_in[5];
  const float* Wq     = (const float*)d_in[6];
  const float* Wk     = (const float*)d_in[7];
  const float* Wv     = (const float*)d_in[8];
  const float* bv     = (const float*)d_in[9];
  const float* W_ih   = (const float*)d_in[10];
  const float* W_hh   = (const float*)d_in[11];
  const float* b_ih   = (const float*)d_in[12];
  const float* b_hh   = (const float*)d_in[13];

  // Outputs, flat in return order: h_mem, slots, cum_feats, delta_t, filled.
  float* out       = (float*)d_out;
  float* h_out     = out;
  float* slots_out = h_out     + (size_t)B_SZ * H_SZ;
  float* cum_out   = slots_out + (size_t)B_SZ * N_SLOT * D_SZ;
  float* dt_out    = cum_out   + (size_t)B_SZ * N_SLOT * D_SZ;
  float* fil_out   = dt_out    + (size_t)B_SZ * N_SLOT;

  // Workspace layout:
  //   [0)                bf16 Wcat   (4H x 768)            ~1.5 MB
  //   [wcat_bytes)       f32  Wd     (4H), f32 bias (4H)      8 KB
  //   [xbf_off)          bf16 mem_seq x-part (B x N x 512) ~134 MB (optional)
  size_t wcat_bytes = (size_t)4 * H_SZ * KCAT * sizeof(bf16_t);
  size_t misc_bytes = (size_t)4 * H_SZ * 2 * sizeof(float);
  size_t xbf_off    = (wcat_bytes + misc_bytes + 255) & ~(size_t)255;
  size_t xbf_bytes  = (size_t)B_SZ * N_SLOT * 512 * sizeof(bf16_t);

  bf16_t* Wcat = (bf16_t*)d_ws;
  float*  Wd   = (float*)((char*)d_ws + wcat_bytes);
  float*  bias = Wd + 4 * H_SZ;
  bf16_t* Xbf  = (ws_size >= xbf_off + xbf_bytes)
                   ? (bf16_t*)((char*)d_ws + xbf_off) : nullptr;

  prep_weights <<<4 * H_SZ, 256, 0, stream>>>(W_ih, W_hh, b_ih, b_hh, Wcat, Wd, bias);
  select_kernel<<<B_SZ,     256, 0, stream>>>(x_t, slots, cum, dt, filled,
                                              Wq, Wk, Wv, bv,
                                              slots_out, cum_out, dt_out, fil_out,
                                              Xbf);
  if (Xbf) {
    lstm_kernel<true><<<B_SZ / 16, 256, 0, stream>>>(
        slots_out, cum_out, Xbf, dt_out, Wcat, Wd, bias, h_out);
  } else {
    lstm_kernel<false><<<B_SZ / 16, 256, 0, stream>>>(
        slots_out, cum_out, nullptr, dt_out, Wcat, Wd, bias, h_out);
  }
}
